// RecurrentAttention_89369679495767
// MI455X (gfx1250) — matrix operand order repaired and checked by reading, not compiled, not run on hardware
//
#include <hip/hip_runtime.h>
#include <math.h>

// ---------------------------------------------------------------------------
// RecurrentAttention (LSTM + Luong attention, input feeding) for MI455X
// B=32, T=128, S=128, Din=H=C=1024.
// bf16 WMMA (v_wmma_f32_16x16x32_bf16) for all GEMMs, fp32 state & softmax.
// 4 kernels per time step, software-pipelined K loops, branch-free segments.
// ---------------------------------------------------------------------------

typedef __bf16 bf16_t;
typedef bf16_t v16bf __attribute__((ext_vector_type(16)));
typedef float  v8f   __attribute__((ext_vector_type(8)));

#define NB   32      // batch
#define NT   128     // time steps
#define NS   128     // context length
#define DIN  1024
#define NH   1024
#define NC   1024
#define G4H  4096    // 4*H

// ---------------- bf16 conversion (round to nearest even) -------------------
static __device__ __forceinline__ unsigned short f32_to_bf16_bits(float f) {
  unsigned u = __float_as_uint(f);
  unsigned r = u + 0x7FFFu + ((u >> 16) & 1u);
  return (unsigned short)(r >> 16);
}

static __device__ __forceinline__ float sigmoidf_(float x) {
  return 1.0f / (1.0f + __expf(-x));
}

// ---------------- WMMA fragment loaders (ISA 7.12.2 layouts) ----------------
// A fragment, 16-bit, 16x32 (MxK), from row-major bf16 matrix (stride lda):
//   lanes 0-15 : row M=lane,    v0..v3 = K 0..7,  v4..v7 = K 16..23
//   lanes 16-31: row M=lane-16, v0..v3 = K 8..15, v4..v7 = K 24..31
static __device__ __forceinline__ v16bf load_a_frag(const unsigned short* base,
                                                    long lda, int m0, int k0,
                                                    int lane) {
  int row = m0 + (lane & 15);
  const unsigned short* p = base + (long)row * lda + k0 + ((lane >> 4) << 3);
  union { uint4 u[2]; v16bf v; } x;
  x.u[0] = *(const uint4*)(p);
  x.u[1] = *(const uint4*)(p + 16);
  return x.v;
}

// B fragment, 16-bit, 32x16 (KxN), B[k][n] = W[n0+n][k0+k] from row-major W:
// lanes 0-15: K 0..7,16..23; lanes 16-31: K 8..15,24..31 (same as A).
static __device__ __forceinline__ v16bf load_b_frag(const unsigned short* w,
                                                    long ldw, int n0, int k0,
                                                    int lane) {
  int nn = n0 + (lane & 15);
  const unsigned short* p = w + (long)nn * ldw + k0 + ((lane >> 4) << 3);
  union { uint4 u[2]; v16bf v; } x;
  x.u[0] = *(const uint4*)(p);
  x.u[1] = *(const uint4*)(p + 16);
  return x.v;
}

static __device__ __forceinline__ v8f wmma_bf16(v16bf a, v16bf b, v8f c) {
  return __builtin_amdgcn_wmma_f32_16x16x32_bf16(false, a, false, b, (short)0,
                                                 c, false, false);
}

// Pipelined GEMM segment: acc[0:2] (M=0..15, 16..31) x W-strip n0, over
// nk 32-wide K chunks.  Fragments for kt+1 are loaded before the WMMAs of kt
// so the s_wait overlaps with matrix math.
static __device__ __forceinline__ void gemm_seg(
    const unsigned short* __restrict__ a, long lda,
    const unsigned short* __restrict__ w, long ldw, int n0, int wk0, int nk,
    int lane, v8f& acc0, v8f& acc1) {
  v16bf bf = load_b_frag(w, ldw, n0, wk0, lane);
  v16bf a0 = load_a_frag(a, lda, 0, 0, lane);
  v16bf a1 = load_a_frag(a, lda, 16, 0, lane);
  for (int kt = 1; kt < nk; ++kt) {
    int k0 = kt * 32;
    v16bf bfn = load_b_frag(w, ldw, n0, wk0 + k0, lane);
    v16bf a0n = load_a_frag(a, lda, 0, k0, lane);
    v16bf a1n = load_a_frag(a, lda, 16, k0, lane);
    acc0 = wmma_bf16(a0, bf, acc0);
    acc1 = wmma_bf16(a1, bf, acc1);
    bf = bfn; a0 = a0n; a1 = a1n;
  }
  acc0 = wmma_bf16(a0, bf, acc0);
  acc1 = wmma_bf16(a1, bf, acc1);
}

// ---------------------------- prep kernels ----------------------------------
__global__ void cvt_bf16_kernel(const float* __restrict__ src,
                                unsigned short* __restrict__ dst, int n) {
  int i = blockIdx.x * blockDim.x + threadIdx.x;
  if (i < n) dst[i] = f32_to_bf16_bits(src[i]);
}

__global__ void init_state_kernel(float* __restrict__ c,
                                  unsigned short* __restrict__ hbf,
                                  unsigned short* __restrict__ attnbf) {
  int i = blockIdx.x * blockDim.x + threadIdx.x;  // 32768 threads
  c[i] = 0.0f; hbf[i] = 0; attnbf[i] = 0;
}

// ------------------- fused gates GEMM + LSTM cell ----------------------------
// Block (4 waves) owns hidden cols [hc0, hc0+16).  Wave w computes the gate-w
// strip (rows w*1024+hc0 of [W_ih|W_hh]) over K = [x_t | attn | h] (3072),
// parks the 32x16 fp32 tile in LDS, then the block applies the LSTM pointwise:
//   c' = sig(f)*c + sig(i)*tanh(g);  h' = sig(o)*tanh(c')
__global__ __launch_bounds__(128) void gates_cell_kernel(
    const unsigned short* __restrict__ xbf, int t,
    const unsigned short* __restrict__ attnbf,
    const unsigned short* __restrict__ hbf_prev,
    const unsigned short* __restrict__ wih,   // [4096,2048] bf16 row-major
    const unsigned short* __restrict__ whh,   // [4096,1024]
    const float* __restrict__ b_ih, const float* __restrict__ b_hh,
    float* __restrict__ c, unsigned short* __restrict__ hbf_next) {
  __shared__ float gsm[4][NB][16];            // [gate][batch][col] = 8 KB
  int tid = threadIdx.x;
  int lane = tid & 31, w = tid >> 5;          // gate group 0..3 (i,f,g,o)
  int hc0 = blockIdx.x * 16;                  // hidden col group
  int n0 = w * 1024 + hc0;                    // row strip in W
  const unsigned short* xb = xbf + (long)t * DIN;  // row stride NT*DIN

  v8f acc0 = {}; v8f acc1 = {};
  gemm_seg(xb,       (long)NT * DIN, wih, 2048, n0, 0,    32, lane, acc0, acc1);
  gemm_seg(attnbf,   NH,             wih, 2048, n0, 1024, 32, lane, acc0, acc1);
  gemm_seg(hbf_prev, NH,             whh, 1024, n0, 0,    32, lane, acc0, acc1);

  int col = lane & 15;
  int mb = (lane >> 4) * 8;
  #pragma unroll
  for (int j = 0; j < 8; ++j) {
    gsm[w][mb + j][col]      = acc0[j];
    gsm[w][mb + j + 16][col] = acc1[j];
  }
  __syncthreads();

  // 512 cell elements (32 batches x 16 cols), 4 per thread
  #pragma unroll
  for (int r = 0; r < 4; ++r) {
    int e = tid * 4 + r;
    int b = e >> 4, cc = e & 15;
    int n = hc0 + cc;
    float bi = b_ih[n] + b_hh[n];
    float bf_ = b_ih[n + 1024] + b_hh[n + 1024];
    float bg = b_ih[n + 2048] + b_hh[n + 2048];
    float bo = b_ih[n + 3072] + b_hh[n + 3072];
    float gi = gsm[0][b][cc] + bi;
    float gf = gsm[1][b][cc] + bf_;
    float gg = gsm[2][b][cc] + bg;
    float go = gsm[3][b][cc] + bo;
    long idx = (long)b * NH + n;
    float cn = sigmoidf_(gf) * c[idx] + sigmoidf_(gi) * tanhf(gg);
    float hn = sigmoidf_(go) * tanhf(cn);
    c[idx] = cn;
    hbf_next[idx] = f32_to_bf16_bits(hn);
  }
}

// ---------------------------- q = h @ Wq^T ----------------------------------
__global__ __launch_bounds__(128) void qproj_kernel(
    const unsigned short* __restrict__ hbf,
    const unsigned short* __restrict__ wq,  // [1024,1024] bf16 row-major
    float* __restrict__ q) {
  int wave = (blockIdx.x * blockDim.x + threadIdx.x) >> 5;
  int lane = threadIdx.x & 31;
  int n0 = wave * 16;
  v8f acc0 = {}; v8f acc1 = {};
  gemm_seg(hbf, NH, wq, 1024, n0, 0, 32, lane, acc0, acc1);
  int n = n0 + (lane & 15);
  int mb = (lane >> 4) * 8;
  #pragma unroll
  for (int j = 0; j < 8; ++j) {
    q[(long)(mb + j) * NC + n]      = acc0[j];
    q[(long)(mb + j + 16) * NC + n] = acc1[j];
  }
}

// ---- fused attention: scores -> softmax -> ctx (one block per batch) -------
__global__ __launch_bounds__(256) void attend_kernel(
    const float* __restrict__ q, const float* __restrict__ context,
    unsigned short* __restrict__ ctxbf) {
  __shared__ float sm[NS];                 // scores, then alpha
  int b = blockIdx.x;
  int tid = threadIdx.x, lane = tid & 31, w = tid >> 5;  // 8 waves
  const float* qv = q + (long)b * NC;
  const float* cb = context + (long)b * NS * NC;

  // scores: wave w owns s = w*16 .. w*16+15 (coalesced dot over C)
  for (int si = 0; si < 16; ++si) {
    int s = w * 16 + si;
    const float* cv = cb + (long)s * NC;
    float sum = 0.0f;
    for (int cidx = lane; cidx < NC; cidx += 32) sum += qv[cidx] * cv[cidx];
    #pragma unroll
    for (int off = 16; off > 0; off >>= 1) sum += __shfl_xor(sum, off, 32);
    if (lane == 0) sm[s] = sum;
  }
  __syncthreads();

  // softmax over 128 scores (every thread redundantly; LDS broadcast reads)
  float mm = sm[0];
  for (int s = 1; s < NS; ++s) mm = fmaxf(mm, sm[s]);
  float denom = 0.0f;
  for (int s = 0; s < NS; ++s) denom += __expf(sm[s] - mm);
  float inv = 1.0f / denom;
  __syncthreads();
  if (tid < NS) sm[tid] = __expf(sm[tid] - mm) * inv;  // alpha
  __syncthreads();

  // ctx[b,c] = sum_s alpha[s] * context[b,s,c]  (coalesced over c)
  for (int ci = tid; ci < NC; ci += 256) {
    float acc = 0.0f;
    for (int s = 0; s < NS; ++s) acc += sm[s] * cb[(long)s * NC + ci];
    ctxbf[(long)b * NC + ci] = f32_to_bf16_bits(acc);
  }
}

// --------- out = tanh([ctx|h] @ Wout^T + b_out); stage attn for t+1 ---------
__global__ __launch_bounds__(128) void outproj_kernel(
    const unsigned short* __restrict__ ctxbf,
    const unsigned short* __restrict__ hbf,
    const unsigned short* __restrict__ wout,  // [1024,2048] bf16 row-major
    const float* __restrict__ b_out, int t,
    float* __restrict__ out,                  // [B,T,H]
    unsigned short* __restrict__ attnbf) {
  int wave = (blockIdx.x * blockDim.x + threadIdx.x) >> 5;
  int lane = threadIdx.x & 31;
  int n0 = wave * 16;
  v8f acc0 = {}; v8f acc1 = {};
  gemm_seg(ctxbf, NC, wout, 2048, n0, 0,    32, lane, acc0, acc1);
  gemm_seg(hbf,   NH, wout, 2048, n0, 1024, 32, lane, acc0, acc1);
  int n = n0 + (lane & 15);
  float bias = b_out[n];
  int mb = (lane >> 4) * 8;
  #pragma unroll
  for (int j = 0; j < 8; ++j) {
    float v0 = tanhf(acc0[j] + bias);
    float v1 = tanhf(acc1[j] + bias);
    int b0 = mb + j, b1 = mb + j + 16;
    out[((long)b0 * NT + t) * NH + n] = v0;
    out[((long)b1 * NT + t) * NH + n] = v1;
    attnbf[(long)b0 * NH + n] = f32_to_bf16_bits(v0);
    attnbf[(long)b1 * NH + n] = f32_to_bf16_bits(v1);
  }
}

// ---------------------------------------------------------------------------
extern "C" void kernel_launch(void* const* d_in, const int* in_sizes, int n_in,
                              void* d_out, int out_size, void* d_ws,
                              size_t ws_size, hipStream_t stream) {
  (void)in_sizes; (void)n_in; (void)out_size; (void)ws_size;
  const float* x       = (const float*)d_in[0];  // [32,128,1024]
  const float* context = (const float*)d_in[1];  // [32,128,1024]
  const float* W_ih    = (const float*)d_in[2];  // [4096,2048]
  const float* W_hh    = (const float*)d_in[3];  // [4096,1024]
  const float* b_ih    = (const float*)d_in[4];  // [4096]
  const float* b_hh    = (const float*)d_in[5];  // [4096]
  const float* Wq      = (const float*)d_in[6];  // [1024,1024]
  const float* Wout    = (const float*)d_in[7];  // [1024,2048]
  const float* b_out   = (const float*)d_in[8];  // [1024]
  float* out = (float*)d_out;                    // [32,128,1024]

  // ---- workspace layout (256B-aligned) ----
  char* ws = (char*)d_ws;
  size_t off = 0;
  auto alloc = [&](size_t bytes) {
    char* p = ws + off;
    off += (bytes + 255) & ~(size_t)255;
    return p;
  };
  unsigned short* wihbf  = (unsigned short*)alloc((size_t)G4H * 2048 * 2);
  unsigned short* whhbf  = (unsigned short*)alloc((size_t)G4H * 1024 * 2);
  unsigned short* wqbf   = (unsigned short*)alloc((size_t)NC * NH * 2);
  unsigned short* woutbf = (unsigned short*)alloc((size_t)NH * 2048 * 2);
  unsigned short* xbf    = (unsigned short*)alloc((size_t)NB * NT * DIN * 2);
  float*          cstate = (float*)alloc((size_t)NB * NH * 4);
  float*          q      = (float*)alloc((size_t)NB * NC * 4);
  unsigned short* hbf_a  = (unsigned short*)alloc((size_t)NB * NH * 2);
  unsigned short* hbf_b  = (unsigned short*)alloc((size_t)NB * NH * 2);
  unsigned short* attnbf = (unsigned short*)alloc((size_t)NB * NH * 2);
  unsigned short* ctxbf  = (unsigned short*)alloc((size_t)NB * NC * 2);

  // ---- one-time prep: bf16 conversions + state init ----
  auto cvt = [&](const float* src, unsigned short* dst, int n) {
    cvt_bf16_kernel<<<(n + 255) / 256, 256, 0, stream>>>(src, dst, n);
  };
  cvt(W_ih, wihbf, G4H * 2048);
  cvt(W_hh, whhbf, G4H * 1024);
  cvt(Wq,   wqbf,  NC * NH);
  cvt(Wout, woutbf, NH * 2048);
  cvt(x,    xbf,   NB * NT * DIN);
  init_state_kernel<<<128, 256, 0, stream>>>(cstate, hbf_a, attnbf);

  // ---- sequential scan over time: 4 kernels per step ----
  for (int t = 0; t < NT; ++t) {
    unsigned short* hcur = (t & 1) ? hbf_b : hbf_a;  // h_{t-1}
    unsigned short* hnxt = (t & 1) ? hbf_a : hbf_b;  // h_t
    gates_cell_kernel<<<64, 128, 0, stream>>>(xbf, t, attnbf, hcur, wihbf,
                                              whhbf, b_ih, b_hh, cstate, hnxt);
    qproj_kernel<<<16, 128, 0, stream>>>(hnxt, wqbf, q);
    attend_kernel<<<NB, 256, 0, stream>>>(q, context, ctxbf);
    outproj_kernel<<<16, 128, 0, stream>>>(ctxbf, hnxt, woutbf, b_out, t, out,
                                           attnbf);
  }
}
